// LineRWKVBlock_56564719288736
// MI455X (gfx1250) — matrix operand order repaired and checked by reading, not compiled, not run on hardware
//
#include <hip/hip_runtime.h>
#include <hip/hip_bf16.h>
#include <cmath>

// Problem constants (from reference)
#define DIM   1024
#define BATCH 16
#define WIDTH 2048
#define MROWS (BATCH * WIDTH)   // 32768
#define DECAYF 0.99f

// Scan chunking: 16 chunks of 128 -> 262144-way parallel scan passes
#define CHUNK  128
#define NCHUNK (WIDTH / CHUNK)  // 16

// LDS tile stride in bf16 elements: 56*2=112B rows -> 16B aligned, 28-bank stride
#define LDS_STRIDE 56
#define KSTEPS (DIM / 32)       // 32 k-iterations

typedef __bf16          bf16x16  __attribute__((ext_vector_type(16)));
typedef unsigned short  ushort16 __attribute__((ext_vector_type(16)));
typedef unsigned short  ushort8  __attribute__((ext_vector_type(8)));
typedef float           f32x8    __attribute__((ext_vector_type(8)));
typedef int             int4v    __attribute__((ext_vector_type(4)));

union FragU { ushort16 v; ushort8 h[2]; };

__device__ __forceinline__ unsigned short f2bf(float f) {
  // round-to-nearest-even f32 -> bf16
  unsigned u = __float_as_uint(f);
  u += 0x7FFFu + ((u >> 16) & 1u);
  return (unsigned short)(u >> 16);
}

// ---------------------------------------------------------------------------
// CDNA5 async global->LDS copy (ASYNCcnt-tracked, no VGPR staging) with a
// portable fallback.  Copies 32 bytes (16 bf16).
// Builtin signature (per hipcc diagnostic): arg0 = int4* in global AS.
// ---------------------------------------------------------------------------
__device__ __forceinline__ void cp_async_32B(const unsigned short* g,
                                             unsigned short* l) {
#if defined(__gfx1250__) && __has_builtin(__builtin_amdgcn_global_load_async_to_lds_b128)
  typedef __attribute__((address_space(1))) int4v* gp_t;
  typedef __attribute__((address_space(3))) int4v* lp_t;
  __builtin_amdgcn_global_load_async_to_lds_b128((gp_t)g,       (lp_t)l,       0, 0);
  __builtin_amdgcn_global_load_async_to_lds_b128((gp_t)(g + 8), (lp_t)(l + 8), 0, 0);
#else
  ((ushort8*)l)[0] = ((const ushort8*)g)[0];
  ((ushort8*)l)[1] = ((const ushort8*)g)[1];
#endif
}

__device__ __forceinline__ void wait_async_zero() {
#if defined(__gfx1250__) && __has_builtin(__builtin_amdgcn_s_wait_asynccnt)
  __builtin_amdgcn_s_wait_asynccnt(0);
#elif defined(__gfx1250__)
  asm volatile("s_wait_asynccnt 0x0" ::: "memory");
#endif
}

// ---------------------------------------------------------------------------
// A-tile prefetch: 16 f32 in registers, converted to bf16 on LDS store.
// ---------------------------------------------------------------------------
struct APre { float4 f[4]; };

__device__ __forceinline__ APre a_load(const float* __restrict__ X,
                                       int row, int colbase) {
  const float4* p = (const float4*)(X + (size_t)row * DIM + colbase);
  APre a;
#pragma unroll
  for (int j = 0; j < 4; ++j) a.f[j] = p[j];
  return a;
}

__device__ __forceinline__ void a_store(unsigned short* sA, int r, int c,
                                        const APre& a) {
  unsigned short tmp[16];
#pragma unroll
  for (int j = 0; j < 4; ++j) {
    tmp[4 * j + 0] = f2bf(a.f[j].x);
    tmp[4 * j + 1] = f2bf(a.f[j].y);
    tmp[4 * j + 2] = f2bf(a.f[j].z);
    tmp[4 * j + 3] = f2bf(a.f[j].w);
  }
  ushort8* dst = (ushort8*)&sA[r * LDS_STRIDE + c];
  dst[0] = *(const ushort8*)&tmp[0];
  dst[1] = *(const ushort8*)&tmp[8];
}

// ---------------------------------------------------------------------------
// Pre-pass: transpose + convert 1024x1024 f32 weight into bf16 [n][k].
// ---------------------------------------------------------------------------
__global__ __launch_bounds__(256) void k_prep_wt(const float* __restrict__ src,
                                                 unsigned short* __restrict__ dst) {
  __shared__ float tile[32][33];
  const int bk = blockIdx.x * 32;
  const int bn = blockIdx.y * 32;
  const int tx = threadIdx.x;
  const int ty = threadIdx.y;
#pragma unroll
  for (int j = 0; j < 32; j += 8)
    tile[ty + j][tx] = src[(size_t)(bk + ty + j) * DIM + bn + tx];
  __syncthreads();
#pragma unroll
  for (int j = 0; j < 32; j += 8)
    dst[(size_t)(bn + ty + j) * DIM + bk + tx] = f2bf(tile[tx][ty + j]);
}

// ---------------------------------------------------------------------------
// WMMA fragment helpers (CDNA5 wave32 16-bit layouts)
// ---------------------------------------------------------------------------
__device__ __forceinline__ ushort16 load_a_frag(const unsigned short* sA,
                                                int row, int lane) {
  const int r  = row + (lane & 15);
  const int kb = (lane & 16) ? 8 : 0;
  const ushort8* p = (const ushort8*)&sA[r * LDS_STRIDE + kb];
  FragU f;
  f.h[0] = p[0];   // K = kb .. kb+7
  f.h[1] = p[2];   // K = 16+kb .. 16+kb+7
  return f.v;
}

__device__ __forceinline__ ushort16 load_b_frag(const unsigned short* sB,
                                                int col, int lane) {
  const int n  = col + (lane & 15);
  const int kb = (lane & 16) ? 8 : 0;
  const ushort8* p = (const ushort8*)&sB[n * LDS_STRIDE + kb];
  FragU f;
  f.h[0] = p[0];
  f.h[1] = p[2];
  return f.v;
}

#define WMMA_BF16(a, b, c)                                            \
  __builtin_amdgcn_wmma_f32_16x16x32_bf16(                            \
      false, __builtin_bit_cast(bf16x16, (a)),                        \
      false, __builtin_bit_cast(bf16x16, (b)),                        \
      (short)0, (c), false, false)

// ---------------------------------------------------------------------------
// Stage 1: fused dual GEMM, double-buffered, single barrier per k-step.
//   V = X*Wv ; R = sigmoid(X*Wr)
// ---------------------------------------------------------------------------
__global__ __launch_bounds__(128) void k_gemm_vr(const float* __restrict__ X,
                                                 const unsigned short* __restrict__ BvT,
                                                 const unsigned short* __restrict__ BrT,
                                                 float* __restrict__ V,
                                                 float* __restrict__ R) {
  __shared__ unsigned short sA [2][64 * LDS_STRIDE];
  __shared__ unsigned short sBv[2][64 * LDS_STRIDE];
  __shared__ unsigned short sBr[2][64 * LDS_STRIDE];

  const int t    = threadIdx.x;
  const int lane = t & 31;
  const int wave = t >> 5;
  const int bm   = blockIdx.x * 64;
  const int bn   = blockIdx.y * 64;
  const int msub = (wave & 1) * 32;
  const int nsub = (wave >> 1) * 32;

  const int ar = t >> 1;          // staging row 0..63
  const int ac = (t & 1) * 16;    // staging k-col base (0/16)

  f32x8 accV[2][2] = {};
  f32x8 accR[2][2] = {};

  // prologue: fill buffer 0
  {
    APre a = a_load(X, bm + ar, ac);
    cp_async_32B(BvT + (size_t)(bn + ar) * DIM + ac, &sBv[0][ar * LDS_STRIDE + ac]);
    cp_async_32B(BrT + (size_t)(bn + ar) * DIM + ac, &sBr[0][ar * LDS_STRIDE + ac]);
    a_store(&sA[0][0], ar, ac, a);
  }

  for (int i = 0; i < KSTEPS; ++i) {
    const int cur = i & 1;
    const int nxt = cur ^ 1;
    wait_async_zero();      // our async fills of buffer `cur` are done
    __syncthreads();        // everyone's fills visible; prev compute finished

    const bool pre = (i + 1) < KSTEPS;
    APre a;
    if (pre) {
      const int kk = (i + 1) * 32;
      a = a_load(X, bm + ar, kk + ac);                                   // HBM latency
      cp_async_32B(BvT + (size_t)(bn + ar) * DIM + kk + ac,              // overlaps
                   &sBv[nxt][ar * LDS_STRIDE + ac]);                     // the WMMAs
      cp_async_32B(BrT + (size_t)(bn + ar) * DIM + kk + ac,              // below
                   &sBr[nxt][ar * LDS_STRIDE + ac]);
    }

    ushort16 aF[2], bvF[2], brF[2];
#pragma unroll
    for (int mi = 0; mi < 2; ++mi) aF[mi]  = load_a_frag(sA[cur],  msub + 16 * mi, lane);
#pragma unroll
    for (int ni = 0; ni < 2; ++ni) bvF[ni] = load_b_frag(sBv[cur], nsub + 16 * ni, lane);
#pragma unroll
    for (int ni = 0; ni < 2; ++ni) brF[ni] = load_b_frag(sBr[cur], nsub + 16 * ni, lane);

#pragma unroll
    for (int mi = 0; mi < 2; ++mi)
#pragma unroll
      for (int ni = 0; ni < 2; ++ni) {
        accV[mi][ni] = WMMA_BF16(aF[mi], bvF[ni], accV[mi][ni]);
        accR[mi][ni] = WMMA_BF16(aF[mi], brF[ni], accR[mi][ni]);
      }

    if (pre) a_store(&sA[nxt][0], ar, ac, a);   // convert after compute issued
  }

  // epilogue (C/D layout: lanes 0-15 -> M 0..7, lanes 16-31 -> M 8..15)
#pragma unroll
  for (int mi = 0; mi < 2; ++mi)
#pragma unroll
    for (int ni = 0; ni < 2; ++ni) {
      const int n  = bn + nsub + 16 * ni + (lane & 15);
      const int m0 = bm + msub + 16 * mi + ((lane >> 4) & 1) * 8;
#pragma unroll
      for (int i = 0; i < 8; ++i) {
        const size_t o = (size_t)(m0 + i) * DIM + n;
        V[o] = accV[mi][ni][i];
        // fast sigmoid: v_exp_f32 + v_rcp_f32 (no precise-division expansion)
        R[o] = __builtin_amdgcn_rcpf(1.0f + __expf(-accR[mi][ni][i]));
      }
    }
}

// ---------------------------------------------------------------------------
// Stage 2a: per-chunk local scan with zero init.
// ---------------------------------------------------------------------------
__global__ __launch_bounds__(256) void k_scan_local(const float* __restrict__ V,
                                                    float* __restrict__ local) {
  const int d = blockIdx.x * 256 + threadIdx.x;
  const int c = blockIdx.y;
  const int b = blockIdx.z;
  const size_t base = ((size_t)b * WIDTH + (size_t)c * CHUNK) * DIM + d;
  float s = 0.0f;
#pragma unroll 4
  for (int i = 0; i < CHUNK; ++i)
    s = DECAYF * s + (1.0f - DECAYF) * V[base + (size_t)i * DIM];
  local[((size_t)b * NCHUNK + c) * DIM + d] = s;
}

// ---------------------------------------------------------------------------
// Stage 2b: combine chunk carries (16 steps) + emit exact new_state.
// ---------------------------------------------------------------------------
__global__ __launch_bounds__(256) void k_scan_starts(const float* __restrict__ state0,
                                                     const float* __restrict__ local,
                                                     float* __restrict__ starts,
                                                     float* __restrict__ new_state,
                                                     float decayL) {
  const int idx = blockIdx.x * 256 + threadIdx.x;
  const int b = idx / DIM;
  const int d = idx % DIM;
  float s = state0[(size_t)b * DIM + d];
#pragma unroll
  for (int c = 0; c < NCHUNK; ++c) {
    const size_t o = ((size_t)b * NCHUNK + c) * DIM + d;
    starts[o] = s;
    s = decayL * s + local[o];
  }
  new_state[(size_t)b * DIM + d] = s;
}

// ---------------------------------------------------------------------------
// Stage 2c: replay chunks with true initial state; y_mid = x + r*s in place.
// ---------------------------------------------------------------------------
__global__ __launch_bounds__(256) void k_scan_apply(const float* __restrict__ X,
                                                    const float* __restrict__ R,
                                                    const float* __restrict__ starts,
                                                    float* __restrict__ VY) {
  const int d = blockIdx.x * 256 + threadIdx.x;
  const int c = blockIdx.y;
  const int b = blockIdx.z;
  const size_t base = ((size_t)b * WIDTH + (size_t)c * CHUNK) * DIM + d;
  float s = starts[((size_t)b * NCHUNK + c) * DIM + d];
#pragma unroll 4
  for (int i = 0; i < CHUNK; ++i) {
    const size_t o = base + (size_t)i * DIM;
    s = DECAYF * s + (1.0f - DECAYF) * VY[o];
    VY[o] = X[o] + R[o] * s;
  }
}

// ---------------------------------------------------------------------------
// Stage 3: Y = y_mid + y_mid * Wc  (double-buffered WMMA GEMM + residual)
// ---------------------------------------------------------------------------
__global__ __launch_bounds__(128) void k_gemm_mix(const float* __restrict__ A,
                                                  const unsigned short* __restrict__ BT,
                                                  float* __restrict__ Out) {
  __shared__ unsigned short sA[2][64 * LDS_STRIDE];
  __shared__ unsigned short sB[2][64 * LDS_STRIDE];

  const int t    = threadIdx.x;
  const int lane = t & 31;
  const int wave = t >> 5;
  const int bm   = blockIdx.x * 64;
  const int bn   = blockIdx.y * 64;
  const int msub = (wave & 1) * 32;
  const int nsub = (wave >> 1) * 32;

  const int ar = t >> 1;
  const int ac = (t & 1) * 16;

  f32x8 acc[2][2] = {};

  {
    APre a = a_load(A, bm + ar, ac);
    cp_async_32B(BT + (size_t)(bn + ar) * DIM + ac, &sB[0][ar * LDS_STRIDE + ac]);
    a_store(&sA[0][0], ar, ac, a);
  }

  for (int i = 0; i < KSTEPS; ++i) {
    const int cur = i & 1;
    const int nxt = cur ^ 1;
    wait_async_zero();
    __syncthreads();

    const bool pre = (i + 1) < KSTEPS;
    APre a;
    if (pre) {
      const int kk = (i + 1) * 32;
      a = a_load(A, bm + ar, kk + ac);
      cp_async_32B(BT + (size_t)(bn + ar) * DIM + kk + ac,
                   &sB[nxt][ar * LDS_STRIDE + ac]);
    }

    ushort16 aF[2], bF[2];
#pragma unroll
    for (int mi = 0; mi < 2; ++mi) aF[mi] = load_a_frag(sA[cur], msub + 16 * mi, lane);
#pragma unroll
    for (int ni = 0; ni < 2; ++ni) bF[ni] = load_b_frag(sB[cur], nsub + 16 * ni, lane);

#pragma unroll
    for (int mi = 0; mi < 2; ++mi)
#pragma unroll
      for (int ni = 0; ni < 2; ++ni)
        acc[mi][ni] = WMMA_BF16(aF[mi], bF[ni], acc[mi][ni]);

    if (pre) a_store(&sA[nxt][0], ar, ac, a);
  }

#pragma unroll
  for (int mi = 0; mi < 2; ++mi)
#pragma unroll
    for (int ni = 0; ni < 2; ++ni) {
      const int n  = bn + nsub + 16 * ni + (lane & 15);
      const int m0 = bm + msub + 16 * mi + ((lane >> 4) & 1) * 8;
#pragma unroll
      for (int i = 0; i < 8; ++i) {
        const size_t o = (size_t)(m0 + i) * DIM + n;
        Out[o] = acc[mi][ni][i] + A[o];   // residual
      }
    }
}

// ---------------------------------------------------------------------------
// Host launcher
// ---------------------------------------------------------------------------
extern "C" void kernel_launch(void* const* d_in, const int* in_sizes, int n_in,
                              void* d_out, int out_size, void* d_ws, size_t ws_size,
                              hipStream_t stream) {
  const float* x     = (const float*)d_in[0];
  const float* state = (const float*)d_in[1];
  const float* Wv    = (const float*)d_in[2];
  const float* Wr    = (const float*)d_in[3];
  const float* Wc    = (const float*)d_in[4];

  float* y_out  = (float*)d_out;
  float* ns_out = y_out + (size_t)BATCH * WIDTH * DIM;

  float* v      = (float*)d_ws;                               // 128 MiB (becomes y_mid)
  float* r      = v + (size_t)MROWS * DIM;                    // 128 MiB
  float* local  = r + (size_t)MROWS * DIM;                    // 1 MiB
  float* starts = local + (size_t)BATCH * NCHUNK * DIM;       // 1 MiB
  unsigned short* WvT = (unsigned short*)(starts + (size_t)BATCH * NCHUNK * DIM);
  unsigned short* WrT = WvT + (size_t)DIM * DIM;              // 2 MiB each
  unsigned short* WcT = WrT + (size_t)DIM * DIM;

  {
    dim3 tb(32, 8);
    dim3 tg(DIM / 32, DIM / 32);
    k_prep_wt<<<tg, tb, 0, stream>>>(Wv, WvT);
    k_prep_wt<<<tg, tb, 0, stream>>>(Wr, WrT);
    k_prep_wt<<<tg, tb, 0, stream>>>(Wc, WcT);
  }

  k_gemm_vr<<<dim3(MROWS / 64, DIM / 64), 128, 0, stream>>>(x, WvT, WrT, v, r);

  k_scan_local<<<dim3(DIM / 256, NCHUNK, BATCH), 256, 0, stream>>>(v, local);
  const float decayL = (float)pow((double)DECAYF, (double)CHUNK);
  k_scan_starts<<<(BATCH * DIM) / 256, 256, 0, stream>>>(state, local, starts, ns_out, decayL);
  k_scan_apply<<<dim3(DIM / 256, NCHUNK, BATCH), 256, 0, stream>>>(x, r, starts, v);

  k_gemm_mix<<<dim3(MROWS / 64, DIM / 64), 128, 0, stream>>>(v, WcT, y_out);
}
